// Layer_69982197121732
// MI455X (gfx1250) — compile-verified, matrix operand order checked by reading, not run on hardware
//
#include <hip/hip_runtime.h>

typedef __bf16 bf16;
typedef __attribute__((ext_vector_type(16))) __bf16 bf16x16;
typedef __attribute__((ext_vector_type(8)))  float  f32x8;

__device__ __forceinline__ bf16 to_bf16(float x) { return (bf16)x; }

// D = A*B + C, bf16 inputs, f32 accumulate (v_wmma_f32_16x16x32_bf16)
__device__ __forceinline__ f32x8 wmma_bf16(bf16x16 a, bf16x16 b, f32x8 c) {
  return __builtin_amdgcn_wmma_f32_16x16x32_bf16(false, a, false, b, (short)0, c,
                                                 false, false);
}

// ---- VALU cross-lane reductions via DPP16 ROW_XMASK (0x160 + xor mask) ----
template <int CTRL>
__device__ __forceinline__ float dpp_f(float x) {
  return __builtin_bit_cast(
      float, __builtin_amdgcn_update_dpp(0, __builtin_bit_cast(int, x), CTRL,
                                         0xf, 0xf, true));
}
// Reduce across the 16 lanes of each wave32 half-row (matches C-frag halves).
__device__ __forceinline__ float row16_max(float x) {
  x = fmaxf(x, dpp_f<0x161>(x));
  x = fmaxf(x, dpp_f<0x162>(x));
  x = fmaxf(x, dpp_f<0x164>(x));
  x = fmaxf(x, dpp_f<0x168>(x));
  return x;
}
__device__ __forceinline__ float row16_sum(float x) {
  x += dpp_f<0x161>(x);
  x += dpp_f<0x162>(x);
  x += dpp_f<0x164>(x);
  x += dpp_f<0x168>(x);
  return x;
}

__device__ __forceinline__ void wait_asynccnt0() {
#if __has_builtin(__builtin_amdgcn_s_wait_asynccnt)
  __builtin_amdgcn_s_wait_asynccnt(0);
#else
  asm volatile("s_wait_asynccnt 0x0" ::: "memory");
#endif
}

// Async 16-byte global -> LDS copy (GLOBAL_LOAD_ASYNC_TO_LDS_B128, ASYNCcnt).
__device__ __forceinline__ void async_copy16(const bf16* gsrc, bf16* ldst) {
  const unsigned lds_off = (unsigned)(uintptr_t)ldst;  // low 32b = LDS offset
  asm volatile("global_load_async_to_lds_b128 %0, %1, off"
               :: "v"(lds_off), "v"(gsrc) : "memory");
}

// Load a 16x32 16-bit A/B-layout fragment from memory (LDS or global).
// Lane l holds row (l&15); k pattern: (j>>2)*16 + (l>>4)*8 + (j&3)*2.
__device__ __forceinline__ bf16x16 frag_ld(const bf16* p, int rstride) {
  const int lane = threadIdx.x & 31;
  const bf16* rp = p + (lane & 15) * rstride + (lane >> 4) * 8;
  bf16x16 f;
#pragma unroll
  for (int j = 0; j < 8; ++j) {
    const int k = (j >> 2) * 16 + (j & 3) * 2;
    f[2 * j]     = rp[k];
    f[2 * j + 1] = rp[k + 1];
  }
  return f;
}

// ---------------------------------------------------------------------------
// Generic GEMM: C = A[M,K] @ W[N,K]^T   (torch Linear)
// A is f32 (A_F32) or bf16. Workgroup tile 128x64, 8 waves of 32x32.
// Global loads for step k+1 are register-pipelined over step k's WMMAs.
// OUT_MODE 0: bf16 row-major [M,N]
// OUT_MODE 1: head scatter   out[((b*8+h)*4096 + l)*64 + dh]   (Q/K)
// OUT_MODE 2: V^T scatter    out[((b*8+h)*64 + dh)*4096 + l]
// ---------------------------------------------------------------------------
template <bool A_F32, bool RELU, int OUT_MODE>
__global__ __launch_bounds__(256) void gemm_wt(const void* __restrict__ Ap,
                                               const float* __restrict__ W,
                                               bf16* __restrict__ out,
                                               int M, int N, int K) {
  constexpr int BM = 128, BN = 64, BK = 32, AS = BK + 4;  // padded stride 36
  __shared__ __align__(16) bf16 As[BM * AS];
  __shared__ __align__(16) bf16 Ws[BN * AS];
  const int tid  = threadIdx.x;
  const int wave = tid >> 5;
  const int lane = tid & 31;
  const int m0 = blockIdx.x * BM;
  const int n0 = blockIdx.y * BN;
  const int wm = (wave >> 1) * 32;  // 4 waves along M
  const int wn = (wave & 1) * 32;   // 2 waves along N
  const int arow = tid >> 1, ac0 = (tid & 1) * 16;
  const int wrow = tid >> 2, wc0 = (tid & 3) * 8;
  f32x8 acc[2][2] = {};
  float aregf[16];
  bf16  aregb[16];
  float wregf[8];

  auto fetch = [&](int k0) {
    if (A_F32) {
      const float* ag = (const float*)Ap + (size_t)(m0 + arow) * K + k0 + ac0;
#pragma unroll
      for (int i = 0; i < 16; ++i) aregf[i] = ag[i];
    } else {
      const bf16* ag = (const bf16*)Ap + (size_t)(m0 + arow) * K + k0 + ac0;
#pragma unroll
      for (int i = 0; i < 16; ++i) aregb[i] = ag[i];
    }
    const float* wg = W + (size_t)(n0 + wrow) * K + k0 + wc0;
#pragma unroll
    for (int i = 0; i < 8; ++i) wregf[i] = wg[i];
  };

  fetch(0);
  for (int k0 = 0; k0 < K; k0 += BK) {
    __syncthreads();
    {  // publish staged registers to LDS (convert f32 -> bf16 here)
      bf16* la = As + arow * AS + ac0;
#pragma unroll
      for (int i = 0; i < 16; ++i) la[i] = A_F32 ? to_bf16(aregf[i]) : aregb[i];
      bf16* lw = Ws + wrow * AS + wc0;
#pragma unroll
      for (int i = 0; i < 8; ++i) lw[i] = to_bf16(wregf[i]);
    }
    __syncthreads();
    if (k0 + BK < K) fetch(k0 + BK);  // overlap next loads with WMMAs

    bf16x16 af[2], bq[2];
#pragma unroll
    for (int t = 0; t < 2; ++t) af[t] = frag_ld(As + (wm + t * 16) * AS, AS);
#pragma unroll
    for (int t = 0; t < 2; ++t) bq[t] = frag_ld(Ws + (wn + t * 16) * AS, AS);
#pragma unroll
    for (int i = 0; i < 2; ++i)
#pragma unroll
      for (int j = 0; j < 2; ++j) acc[i][j] = wmma_bf16(af[i], bq[j], acc[i][j]);
  }

  // C fragment: VGPR r -> row r + 8*(lane>>4), col = lane&15
  const int cn = lane & 15, mh = (lane >> 4) * 8;
#pragma unroll
  for (int i = 0; i < 2; ++i)
#pragma unroll
    for (int j = 0; j < 2; ++j)
#pragma unroll
      for (int r = 0; r < 8; ++r) {
        const int gm = m0 + wm + i * 16 + mh + r;
        const int gn = n0 + wn + j * 16 + cn;
        float v = acc[i][j][r];
        if (RELU) v = fmaxf(v, 0.0f);
        size_t idx;
        if (OUT_MODE == 0) {
          idx = (size_t)gm * N + gn;
        } else {
          const int b = gm >> 12, l = gm & 4095, h = gn >> 6, dh = gn & 63;
          if (OUT_MODE == 1)
            idx = (((size_t)(b * 8 + h)) << 18) + (size_t)l * 64 + dh;
          else
            idx = (((size_t)(b * 8 + h) * 64 + dh) << 12) + (size_t)l;
        }
        out[idx] = to_bf16(v);
      }
}

// ---------------------------------------------------------------------------
// Full-row GEMM + LayerNorm fusion.  A[M,K] bf16 @ W[512,K]^T, 32 rows/WG,
// 8 waves each own a 64-column slab. LN over the 512 outputs per row.
// EPI 0: bf16 store to outb[row*1024 + 512 + col]   (message half of concat)
// EPI 1: f32 store  outf[row*512 + col] = resid + LN(...)
// ---------------------------------------------------------------------------
template <int EPI>
__global__ __launch_bounds__(256) void gemm_ln(const bf16* __restrict__ A,
                                               const float* __restrict__ W,
                                               const float* __restrict__ gamma,
                                               const float* __restrict__ beta,
                                               const float* __restrict__ resid,
                                               bf16* __restrict__ outb,
                                               float* __restrict__ outf, int K) {
  constexpr int BM = 32, BK = 32, AS = BK + 4;
  __shared__ __align__(16) bf16 As[BM * AS];
  __shared__ __align__(16) bf16 Ws[512 * AS];
  __shared__ float redS[BM], redQ[BM], rmean[BM], rrstd[BM];
  const int tid = threadIdx.x, wave = tid >> 5, lane = tid & 31;
  const int half = lane >> 4, cn = lane & 15;
  const int m0 = blockIdx.x * BM;
  const int wn = wave * 64;
  if (tid < BM) { redS[tid] = 0.0f; redQ[tid] = 0.0f; }
  f32x8 acc[2][4] = {};

  for (int k0 = 0; k0 < K; k0 += BK) {
    __syncthreads();
    {  // A tile 32x32 bf16
      const int row = tid >> 3, c0 = (tid & 7) * 4;
      const bf16* ag = A + (size_t)(m0 + row) * K + k0 + c0;
      bf16* ls = As + row * AS + c0;
      ls[0] = ag[0]; ls[1] = ag[1]; ls[2] = ag[2]; ls[3] = ag[3];
    }
    {  // W tile 512x32 f32 -> bf16 (2 rows per thread)
#pragma unroll
      for (int rr = 0; rr < 2; ++rr) {
        const int row = tid * 2 + rr;
        const float* wg = W + (size_t)row * K + k0;
        bf16* ls = Ws + row * AS;
#pragma unroll
        for (int i = 0; i < 32; ++i) ls[i] = to_bf16(wg[i]);
        if (k0 + BK < K) __builtin_prefetch(wg + BK, 0, 1);
      }
    }
    __syncthreads();
    bf16x16 af0 = frag_ld(As, AS);
    bf16x16 af1 = frag_ld(As + 16 * AS, AS);
#pragma unroll
    for (int j = 0; j < 4; ++j) {
      bf16x16 bq = frag_ld(Ws + (wn + j * 16) * AS, AS);
      acc[0][j] = wmma_bf16(af0, bq, acc[0][j]);
      acc[1][j] = wmma_bf16(af1, bq, acc[1][j]);
    }
  }
  __syncthreads();

  // Row sums / sum-of-squares: DPP reduce over the wave's 64 cols, then LDS.
#pragma unroll
  for (int i = 0; i < 2; ++i)
#pragma unroll
    for (int r = 0; r < 8; ++r) {
      float s = 0.0f, q = 0.0f;
#pragma unroll
      for (int j = 0; j < 4; ++j) { const float v = acc[i][j][r]; s += v; q += v * v; }
      s = row16_sum(s);
      q = row16_sum(q);
      if (cn == 0) {
        const int row = i * 16 + half * 8 + r;
        atomicAdd(&redS[row], s);
        atomicAdd(&redQ[row], q);
      }
    }
  __syncthreads();
  if (tid < BM) {
    const float mu  = redS[tid] * (1.0f / 512.0f);
    const float var = redQ[tid] * (1.0f / 512.0f) - mu * mu;
    rmean[tid] = mu;
    rrstd[tid] = rsqrtf(var + 1e-5f);
  }
  __syncthreads();

#pragma unroll
  for (int i = 0; i < 2; ++i)
#pragma unroll
    for (int j = 0; j < 4; ++j)
#pragma unroll
      for (int r = 0; r < 8; ++r) {
        const int row = i * 16 + half * 8 + r;
        const int col = wn + j * 16 + cn;
        const float y =
            (acc[i][j][r] - rmean[row]) * rrstd[row] * gamma[col] + beta[col];
        const size_t gr = (size_t)(m0 + row);
        if (EPI == 0)
          outb[gr * 1024 + 512 + col] = to_bf16(y);
        else
          outf[gr * 512 + col] = resid[gr * 512 + col] + y;
      }
}

// ---------------------------------------------------------------------------
// Flash attention: per (b,h), Q[L,64] bf16, K[S,64] bf16, Vt[64,S] bf16.
// WG = 128 query rows (8 waves x 16). S loop in chunks of 32.
// K/V tiles double-buffered via GLOBAL_LOAD_ASYNC_TO_LDS_B128 (ASYNCcnt):
// tile i+1 is in flight while tile i feeds the WMMAs; one barrier per iter.
// ---------------------------------------------------------------------------
__global__ __launch_bounds__(256) void attn(const bf16* __restrict__ Q,
                                            const bf16* __restrict__ Kg,
                                            const bf16* __restrict__ Vt,
                                            bf16* __restrict__ O) {
  constexpr int KS = 72, VS = 40, PS = 36;  // padded LDS strides
  constexpr int NIT = 128;                  // 4096 / 32
  __shared__ __align__(16) bf16 Ks[2 * 32 * KS];
  __shared__ __align__(16) bf16 Vs[2 * 64 * VS];
  __shared__ __align__(16) bf16 Ps[8 * 16 * PS];
  const int tid = threadIdx.x, wave = tid >> 5, lane = tid & 31;
  const int half = lane >> 4, cn = lane & 15;
  const int bid = blockIdx.x;
  const int qb = bid & 31, h = (bid >> 5) & 7, b = bid >> 8;
  const int bh = b * 8 + h;
  const bf16* Qp = Q + ((size_t)bh << 18);
  const bf16* Kp = Kg + ((size_t)bh << 18);
  const bf16* Vp = Vt + ((size_t)bh << 18);
  const int q0 = qb * 128 + wave * 16;

  bf16x16 qf[2];
#pragma unroll
  for (int kk = 0; kk < 2; ++kk)
    qf[kk] = frag_ld(Qp + (size_t)q0 * 64 + kk * 32, 64);

  f32x8 o[4] = {};
  float mst[8], lst[8];
#pragma unroll
  for (int r = 0; r < 8; ++r) { mst[r] = -1e30f; lst[r] = 0.0f; }
  bf16* Pw = Ps + wave * 16 * PS;

  // per-thread staging coordinates (16B chunk per async instruction)
  const int krow = tid >> 3, kc0 = (tid & 7) * 8;
  const int vrow = tid >> 2, vc0 = (tid & 3) * 8;

  // prologue: async-load tile 0 into buffer 0
  async_copy16(Kp + (size_t)krow * 64 + kc0, Ks + krow * KS + kc0);
  async_copy16(Vp + ((size_t)vrow << 12) + vc0, Vs + vrow * VS + vc0);

  for (int it = 0; it < NIT; ++it) {
    const int s0 = it * 32;
    wait_asynccnt0();   // own async ops for tile `it` complete
    __syncthreads();    // every wave's tile data visible; prev reads retired
    if (it + 1 < NIT) {  // launch tile it+1 into the other buffer
      const int sn = s0 + 32;
      bf16* Kn = Ks + ((it + 1) & 1) * 32 * KS;
      bf16* Vn = Vs + ((it + 1) & 1) * 64 * VS;
      async_copy16(Kp + (size_t)(sn + krow) * 64 + kc0, Kn + krow * KS + kc0);
      async_copy16(Vp + ((size_t)vrow << 12) + sn + vc0, Vn + vrow * VS + vc0);
    }
    const bf16* Kc = Ks + (it & 1) * 32 * KS;
    const bf16* Vc = Vs + (it & 1) * 64 * VS;

    // scores 16x32: S = Q @ K^T
    f32x8 sc[2];
#pragma unroll
    for (int nt = 0; nt < 2; ++nt) {
      f32x8 z = {};
      bf16x16 kf0 = frag_ld(Kc + nt * 16 * KS, KS);
      bf16x16 kf1 = frag_ld(Kc + nt * 16 * KS + 32, KS);
      z = wmma_bf16(qf[0], kf0, z);
      z = wmma_bf16(qf[1], kf1, z);
      sc[nt] = z;
    }

    // online softmax (DPP row reductions) + emit P into per-wave LDS patch
#pragma unroll
    for (int r = 0; r < 8; ++r) {
      const float s0v = sc[0][r] * 0.125f;
      const float s1v = sc[1][r] * 0.125f;
      const float mx = row16_max(fmaxf(s0v, s1v));
      const float mnew = fmaxf(mst[r], mx);
      const float corr = __expf(mst[r] - mnew);
      const float p0 = __expf(s0v - mnew);
      const float p1 = __expf(s1v - mnew);
      const float rs = row16_sum(p0 + p1);
      lst[r] = lst[r] * corr + rs;
      mst[r] = mnew;
#pragma unroll
      for (int dt = 0; dt < 4; ++dt) o[dt][r] *= corr;
      const int prow = half * 8 + r;
      Pw[prow * PS + cn] = to_bf16(p0);
      Pw[prow * PS + 16 + cn] = to_bf16(p1);
    }

    // O += P @ V   (B fragment = rows of V^T)
    bf16x16 pf = frag_ld(Pw, PS);
#pragma unroll
    for (int dt = 0; dt < 4; ++dt) {
      bf16x16 vf = frag_ld(Vc + dt * 16 * VS, VS);
      o[dt] = wmma_bf16(pf, vf, o[dt]);
    }
  }

  float inv[8];
#pragma unroll
  for (int r = 0; r < 8; ++r) inv[r] = 1.0f / lst[r];
  const size_t rowbase = (size_t)b * 4096 + q0;
#pragma unroll
  for (int dt = 0; dt < 4; ++dt)
#pragma unroll
    for (int r = 0; r < 8; ++r) {
      const size_t gr = rowbase + half * 8 + r;
      const int col = h * 64 + dt * 16 + cn;
      O[gr * 512 + col] = to_bf16(o[dt][r] * inv[r]);
    }
}

// Pack x (f32) into low half of concat buffer as bf16.
__global__ __launch_bounds__(256) void pack_x(const float* __restrict__ x,
                                              bf16* __restrict__ Hb) {
  const int i = blockIdx.x * 256 + threadIdx.x;
  Hb[((size_t)(i >> 9) << 10) + (i & 511)] = to_bf16(x[i]);
}

extern "C" void kernel_launch(void* const* d_in, const int* in_sizes, int n_in,
                              void* d_out, int out_size, void* d_ws,
                              size_t ws_size, hipStream_t stream) {
  (void)in_sizes; (void)n_in; (void)out_size; (void)ws_size;
  const float* x   = (const float*)d_in[0];
  const float* src = (const float*)d_in[1];
  const float* Wq  = (const float*)d_in[2];
  const float* Wk  = (const float*)d_in[3];
  const float* Wv  = (const float*)d_in[4];
  const float* Wm  = (const float*)d_in[5];
  const float* W1  = (const float*)d_in[6];
  const float* W2  = (const float*)d_in[7];
  const float* g1  = (const float*)d_in[8];
  const float* b1  = (const float*)d_in[9];
  const float* g2  = (const float*)d_in[10];
  const float* b2  = (const float*)d_in[11];
  float* out = (float*)d_out;

  char* ws = (char*)d_ws;
  bf16* Qb = (bf16*)(ws);                       // 8 MB  [b][h][l][64]
  bf16* Kb = (bf16*)(ws + ((size_t)8 << 20));   // 8 MB  [b][h][s][64]
  bf16* Vt = (bf16*)(ws + ((size_t)16 << 20));  // 8 MB  [b][h][64][s]
  bf16* Ob = (bf16*)(ws + ((size_t)24 << 20));  // 8 MB  [b*l][512]
  bf16* Hb = (bf16*)(ws + ((size_t)32 << 20));  // 16 MB [b*l][1024] concat
  bf16* Tb = (bf16*)(ws + ((size_t)48 << 20));  // 16 MB [b*l][1024] relu(h@W1^T)

  const int M = 8192;  // B*L
  dim3 blk(256);
  gemm_wt<true, false, 1><<<dim3(64, 8), blk, 0, stream>>>(x,   Wq, Qb, M, 512, 512);
  gemm_wt<true, false, 1><<<dim3(64, 8), blk, 0, stream>>>(src, Wk, Kb, M, 512, 512);
  gemm_wt<true, false, 2><<<dim3(64, 8), blk, 0, stream>>>(src, Wv, Vt, M, 512, 512);
  pack_x<<<dim3(16384), blk, 0, stream>>>(x, Hb);
  attn<<<dim3(512), blk, 0, stream>>>(Qb, Kb, Vt, Ob);
  gemm_ln<0><<<dim3(256), blk, 0, stream>>>(Ob, Wm, g1, b1, nullptr, Hb, nullptr, 512);
  gemm_wt<false, true, 0><<<dim3(64, 16), blk, 0, stream>>>(Hb, W1, Tb, M, 1024, 1024);
  gemm_ln<1><<<dim3(256), blk, 0, stream>>>(Tb, W2, g2, b2, x, nullptr, out, 1024);
}
